// ContextModel_85993835200994
// MI455X (gfx1250) — matrix operand order repaired and checked by reading, not compiled, not run on hardware
//
#include <hip/hip_runtime.h>
#include <math.h>

typedef _Float16 half_t;
typedef __attribute__((ext_vector_type(16))) _Float16 v16h;
typedef __attribute__((ext_vector_type(8)))  float    v8f;

#define C_LAT 192
#define C_PHI 384
#define HID   640
#define HW    16384          // 128*128
#define NPIX  65536          // 4*128*128  (M dimension)
#define LDS_STRIDE 40        // 32 + 8 halves pad; 80B row stride keeps b128 aligned

// ---------------------------------------------------------------------------
// CDNA5 async global->LDS copy (ASYNCcnt-tracked), per CDNA5 ISA + 15.18.
// dst operand is the wave-relative LDS byte offset (= low 32 bits of a
// generic pointer to __shared__, per the LDS aperture truncation rule).
// ---------------------------------------------------------------------------

__device__ __forceinline__ void async_b128(unsigned lds_off, const half_t* g) {
  asm volatile("global_load_async_to_lds_b128 %0, %1, off"
               :: "v"(lds_off), "v"((unsigned long long)(uintptr_t)g)
               : "memory");
}
__device__ __forceinline__ void wait_async0() {
  asm volatile("s_wait_asynccnt 0" ::: "memory");
}
__device__ __forceinline__ unsigned lds_off32(const void* p) {
  return (unsigned)(unsigned long long)p;   // low 32 bits = LDS offset
}

// ---------------------------------------------------------------------------
// Prep kernels
// ---------------------------------------------------------------------------

__global__ void prep_round_kernel(const float* __restrict__ x,
                                  float* __restrict__ out_f32,
                                  half_t* __restrict__ out16) {
  int idx = blockIdx.x * blockDim.x + threadIdx.x;
  if (idx >= 4 * C_LAT * HW) return;
  int b   = idx / (C_LAT * HW);
  int rem = idx - b * (C_LAT * HW);
  int c   = rem >> 14;
  int hw  = rem & (HW - 1);
  float v = rintf(x[idx]);
  out_f32[idx] = v;
  int m = (b << 14) | hw;
  out16[(size_t)m * C_LAT + c] = (half_t)v;
}

__global__ void prep_phi_kernel(const float* __restrict__ phi,
                                half_t* __restrict__ h0) {
  int idx = blockIdx.x * blockDim.x + threadIdx.x;
  if (idx >= 4 * C_PHI * HW) return;
  int b   = idx / (C_PHI * HW);
  int rem = idx - b * (C_PHI * HW);
  int c   = rem >> 14;
  int hw  = rem & (HW - 1);
  int m = (b << 14) | hw;
  h0[(size_t)m * (2 * C_PHI) + C_PHI + c] = (half_t)phi[idx];
}

// mask_w (384,192,5,5): active type-A taps are exactly linear tap ids 0..11.
// Store WB[tap][o][c]  ([N rows][K cols] f16).
__global__ void conv_weight_kernel(const float* __restrict__ mw,
                                   half_t* __restrict__ WB) {
  int idx = blockIdx.x * blockDim.x + threadIdx.x;
  if (idx >= C_PHI * C_LAT * 25) return;
  int o   = idx / (C_LAT * 25);
  int rem = idx - o * (C_LAT * 25);
  int c   = rem / 25;
  int t   = rem - c * 25;
  if (t >= 12) return;
  WB[((size_t)t * C_PHI + o) * C_LAT + c] = (half_t)mw[idx];
}

// w: [K][N] f32 -> wT: [N][K] f16
__global__ void transpose_w_kernel(const float* __restrict__ w,
                                   half_t* __restrict__ wT, int K, int N) {
  int idx = blockIdx.x * blockDim.x + threadIdx.x;
  if (idx >= K * N) return;
  int k = idx / N, n = idx - k * N;
  wT[(size_t)n * K + k] = (half_t)w[idx];
}

// ---------------------------------------------------------------------------
// WMMA fragment helpers (16-bit A 16x32 / B 32x16 VGPR layouts per ISA 7.12.2)
// ---------------------------------------------------------------------------

__device__ __forceinline__ v16h load_frag_A(const half_t* row, int lane) {
  const int kb = (lane < 16) ? 0 : 8;       // pairs 0..7/16..23 | 8..15/24..31
  v16h f;
  ((uint4*)&f)[0] = *(const uint4*)(row + kb);
  ((uint4*)&f)[1] = *(const uint4*)(row + kb + 16);
  return f;
}

__device__ __forceinline__ v16h load_frag_B(const half_t* row, int lane) {
  const int kb = (lane < 16) ? 0 : 8;       // K 0..7/16..23 | 8..15/24..31
  v16h f;
  ((uint4*)&f)[0] = *(const uint4*)(row + kb);
  ((uint4*)&f)[1] = *(const uint4*)(row + kb + 16);
  return f;
}

// ---------------------------------------------------------------------------
// Tiled WMMA GEMM with double-buffered async-LDS pipeline.
// C[M,N] = act(A[M,K] * B^T[N,K] + bias)
// Block tile 128x128, BK=32, 256 threads = 8 waves as 4(m) x 2(n);
// each wave: 2 A-frags x 4 B-frags -> 8 WMMAs / K-chunk, 64 f32 acc regs.
// ---------------------------------------------------------------------------

template <int ACT, int OUTF32>
__global__ __launch_bounds__(256) void gemm_wmma_kernel(
    const half_t* __restrict__ A, int lda,
    const half_t* __restrict__ BT,           // [N][K]
    const float*  __restrict__ bias,
    void* __restrict__ C, int ldc,
    int M, int N, int K) {
  __shared__ __attribute__((aligned(16))) half_t As[2][128 * LDS_STRIDE];
  __shared__ __attribute__((aligned(16))) half_t Bs[2][128 * LDS_STRIDE];

  const int tid  = threadIdx.x;
  const int lane = tid & 31;
  const int wave = tid >> 5;
  const int wm   = wave >> 1;                // 0..3 -> 32-row strip
  const int wn   = wave &  1;                // 0..1 -> 64-col strip
  const int mBase = blockIdx.y * 128;
  const int nBase = blockIdx.x * 128;

  // tile loading: thread -> (row 0..127, 16-half chunk 0..1), 4 async b128
  const int r    = tid >> 1;
  const int coff = (tid & 1) * 16;

  const half_t* Ag = A  + (size_t)(mBase + r) * lda + coff;
  const half_t* Bg = BT + (size_t)(nBase + r) * K  + coff;

  unsigned ldsA[2], ldsB[2];
  ldsA[0] = lds_off32(&As[0][r * LDS_STRIDE + coff]);
  ldsA[1] = lds_off32(&As[1][r * LDS_STRIDE + coff]);
  ldsB[0] = lds_off32(&Bs[0][r * LDS_STRIDE + coff]);
  ldsB[1] = lds_off32(&Bs[1][r * LDS_STRIDE + coff]);

  auto issue = [&](int k0, int buf) {
    async_b128(ldsA[buf],      Ag + k0);
    async_b128(ldsA[buf] + 16, Ag + k0 + 8);
    async_b128(ldsB[buf],      Bg + k0);
    async_b128(ldsB[buf] + 16, Bg + k0 + 8);
  };

  issue(0, 0);
  wait_async0();
  __syncthreads();

  v8f acc[2][4] = {};
  const int nchunk = K >> 5;

  for (int c = 0; c < nchunk; ++c) {
    const int cur = c & 1;
    if (c + 1 < nchunk) issue((c + 1) * 32, cur ^ 1);

    const half_t* a0 = &As[cur][(wm * 32 + (lane & 15)) * LDS_STRIDE];
    v16h af0 = load_frag_A(a0, lane);
    v16h af1 = load_frag_A(a0 + 16 * LDS_STRIDE, lane);
#pragma unroll
    for (int j = 0; j < 4; ++j) {
      const half_t* brow = &Bs[cur][(wn * 64 + j * 16 + (lane & 15)) * LDS_STRIDE];
      v16h bf = load_frag_B(brow, lane);
      acc[0][j] = __builtin_amdgcn_wmma_f32_16x16x32_f16(
          false, af0, false, bf, (short)0, acc[0][j], false, false);
      acc[1][j] = __builtin_amdgcn_wmma_f32_16x16x32_f16(
          false, af1, false, bf, (short)0, acc[1][j], false, false);
    }
    wait_async0();
    __syncthreads();
  }

  const int rowOff = (lane < 16) ? 0 : 8;
#pragma unroll
  for (int i = 0; i < 2; ++i) {
#pragma unroll
    for (int j = 0; j < 4; ++j) {
      const int gc = nBase + wn * 64 + j * 16 + (lane & 15);
      const float bv = bias[gc];
#pragma unroll
      for (int rr = 0; rr < 8; ++rr) {
        const int gr = mBase + wm * 32 + i * 16 + rr + rowOff;
        float v = acc[i][j][rr] + bv;
        if (ACT) v = (v >= 0.f) ? v : 0.01f * v;
        if (OUTF32) ((float*)C)[(size_t)gr * ldc + gc] = v;
        else        ((half_t*)C)[(size_t)gr * ldc + gc] = (half_t)v;
      }
    }
  }
}

// ---------------------------------------------------------------------------
// Masked 5x5 conv as implicit GEMM, same async double-buffered pipeline.
// K-loop flattened to 72 chunks = 12 taps x 6 k-chunks of 32 channels.
// Border rows: EXEC-masked async skip + ds-store of zeros.
// Writes h0[:, 0:384] (ldc = 768).
// ---------------------------------------------------------------------------

__global__ __launch_bounds__(256) void conv_wmma_kernel(
    const half_t* __restrict__ X16,          // [NPIX][192]
    const half_t* __restrict__ WB,           // [12][384][192]
    const float*  __restrict__ bias,         // [384]
    half_t* __restrict__ C, int ldc) {
  __shared__ __attribute__((aligned(16))) half_t As[2][128 * LDS_STRIDE];
  __shared__ __attribute__((aligned(16))) half_t Bs[2][128 * LDS_STRIDE];

  const int tid  = threadIdx.x;
  const int lane = tid & 31;
  const int wave = tid >> 5;
  const int wm   = wave >> 1;
  const int wn   = wave &  1;
  const int mBase = blockIdx.y * 128;
  const int nBase = blockIdx.x * 128;

  const int r    = tid >> 1;
  const int coff = (tid & 1) * 16;

  // spatial coords of this thread's A row
  const int m    = mBase + r;
  const int bimg = m >> 14;
  const int hw   = m & (HW - 1);
  const int h    = hw >> 7;
  const int w    = hw & 127;

  unsigned ldsA[2], ldsB[2];
  ldsA[0] = lds_off32(&As[0][r * LDS_STRIDE + coff]);
  ldsA[1] = lds_off32(&As[1][r * LDS_STRIDE + coff]);
  ldsB[0] = lds_off32(&Bs[0][r * LDS_STRIDE + coff]);
  ldsB[1] = lds_off32(&Bs[1][r * LDS_STRIDE + coff]);

  auto issue = [&](int chunk, int buf) {
    const int tap = chunk / 6;
    const int k0  = (chunk - tap * 6) * 32;
    const int dy = tap / 5 - 2;
    const int dx = tap % 5 - 2;
    const int hh = h + dy;
    const int ww = w + dx;
    const bool valid = (hh >= 0) & (hh < 128) & (ww >= 0) & (ww < 128);
    if (valid) {
      const half_t* Ag =
          X16 + (size_t)((bimg << 14) + (hh << 7) + ww) * C_LAT + k0 + coff;
      async_b128(ldsA[buf],      Ag);
      async_b128(ldsA[buf] + 16, Ag + 8);
    } else {
      uint4 z = {0u, 0u, 0u, 0u};
      *(uint4*)&As[buf][r * LDS_STRIDE + coff]     = z;
      *(uint4*)&As[buf][r * LDS_STRIDE + coff + 8] = z;
    }
    const half_t* Bg =
        WB + ((size_t)tap * C_PHI + nBase + r) * C_LAT + k0 + coff;
    async_b128(ldsB[buf],      Bg);
    async_b128(ldsB[buf] + 16, Bg + 8);
  };

  issue(0, 0);
  wait_async0();
  __syncthreads();

  v8f acc[2][4] = {};
  const int nchunk = 12 * 6;                 // 72

  for (int c = 0; c < nchunk; ++c) {
    const int cur = c & 1;
    if (c + 1 < nchunk) issue(c + 1, cur ^ 1);

    const half_t* a0 = &As[cur][(wm * 32 + (lane & 15)) * LDS_STRIDE];
    v16h af0 = load_frag_A(a0, lane);
    v16h af1 = load_frag_A(a0 + 16 * LDS_STRIDE, lane);
#pragma unroll
    for (int j = 0; j < 4; ++j) {
      const half_t* brow = &Bs[cur][(wn * 64 + j * 16 + (lane & 15)) * LDS_STRIDE];
      v16h bf = load_frag_B(brow, lane);
      acc[0][j] = __builtin_amdgcn_wmma_f32_16x16x32_f16(
          false, af0, false, bf, (short)0, acc[0][j], false, false);
      acc[1][j] = __builtin_amdgcn_wmma_f32_16x16x32_f16(
          false, af1, false, bf, (short)0, acc[1][j], false, false);
    }
    wait_async0();
    __syncthreads();
  }

  const int rowOff = (lane < 16) ? 0 : 8;
#pragma unroll
  for (int i = 0; i < 2; ++i) {
#pragma unroll
    for (int j = 0; j < 4; ++j) {
      const int gc = nBase + wn * 64 + j * 16 + (lane & 15);
      const float bv = bias[gc];
#pragma unroll
      for (int rr = 0; rr < 8; ++rr) {
        const int gr = mBase + wm * 32 + i * 16 + rr + rowOff;
        C[(size_t)gr * ldc + gc] = (half_t)(acc[i][j][rr] + bv);
      }
    }
  }
}

// ---------------------------------------------------------------------------
// Likelihood: norm.cdf((v+.5)/s) - norm.cdf((v-.5)/s), s = max(|.|, 0.11)
// ---------------------------------------------------------------------------

__global__ void likelihood_kernel(const float* __restrict__ cond,   // [NPIX][384]
                                  const float* __restrict__ outv,   // NCHW
                                  float* __restrict__ lik) {
  int idx = blockIdx.x * blockDim.x + threadIdx.x;
  if (idx >= 4 * C_LAT * HW) return;
  int b   = idx / (C_LAT * HW);
  int rem = idx - b * (C_LAT * HW);
  int c   = rem >> 14;
  int hw  = rem & (HW - 1);
  int m = (b << 14) | hw;
  float mean = cond[(size_t)m * (2 * C_LAT) + c];
  float s    = fmaxf(fabsf(cond[(size_t)m * (2 * C_LAT) + C_LAT + c]), 0.11f);
  float v    = outv[idx] - mean;
  float rs   = 0.70710678118654752f / s;     // 1/(s*sqrt(2))
  lik[idx] = 0.5f * (erff((v + 0.5f) * rs) - erff((v - 0.5f) * rs));
}

// ---------------------------------------------------------------------------
// Launch
// ---------------------------------------------------------------------------

extern "C" void kernel_launch(void* const* d_in, const int* in_sizes, int n_in,
                              void* d_out, int out_size, void* d_ws, size_t ws_size,
                              hipStream_t stream) {
  const float* x      = (const float*)d_in[0];
  const float* phi    = (const float*)d_in[1];
  const float* mask_w = (const float*)d_in[2];
  const float* mask_b = (const float*)d_in[3];
  const float* w1     = (const float*)d_in[4];
  const float* b1     = (const float*)d_in[5];
  const float* w2     = (const float*)d_in[6];
  const float* b2     = (const float*)d_in[7];
  const float* w3     = (const float*)d_in[8];
  const float* b3     = (const float*)d_in[9];

  float* out_f32 = (float*)d_out;                   // 4*192*128*128
  float* lik     = out_f32 + (size_t)4 * C_LAT * HW;

  char* ws = (char*)d_ws;
  half_t* OUT16 = (half_t*)(ws + 0);                 //  25,165,824 B
  half_t* H0    = (half_t*)(ws + 25165824);          // 100,663,296 B
  float*  COND  = (float*)(ws + 25165824);           // alias of H0 (dead then)
  half_t* H1    = (half_t*)(ws + 125829120);         //  83,886,080 B
  half_t* H2    = (half_t*)(ws + 209715200);         //  83,886,080 B
  half_t* WB    = (half_t*)(ws + 293601280);         //   1,769,472 B
  half_t* W1T   = (half_t*)(ws + 295370752);         //     983,040 B
  half_t* W2T   = (half_t*)(ws + 296353792);         //     819,200 B
  half_t* W3T   = (half_t*)(ws + 297172992);         //     491,520 B

  const int nx = 4 * C_LAT * HW;                     // 12,582,912
  const int np = 4 * C_PHI * HW;                     // 25,165,824

  prep_round_kernel<<<(nx + 255) / 256, 256, 0, stream>>>(x, out_f32, OUT16);
  prep_phi_kernel<<<(np + 255) / 256, 256, 0, stream>>>(phi, H0);
  conv_weight_kernel<<<(C_PHI * C_LAT * 25 + 255) / 256, 256, 0, stream>>>(mask_w, WB);
  transpose_w_kernel<<<(768 * 640 + 255) / 256, 256, 0, stream>>>(w1, W1T, 768, 640);
  transpose_w_kernel<<<(640 * 640 + 255) / 256, 256, 0, stream>>>(w2, W2T, 640, 640);
  transpose_w_kernel<<<(640 * 384 + 255) / 256, 256, 0, stream>>>(w3, W3T, 640, 384);

  dim3 blk(256);
  // masked conv -> h0[:, 0:384]
  conv_wmma_kernel<<<dim3(C_PHI / 128, NPIX / 128), blk, 0, stream>>>(
      OUT16, WB, mask_b, H0, 2 * C_PHI);
  // layer 1: [M,768] x [768,640], leaky
  gemm_wmma_kernel<1, 0><<<dim3(HID / 128, NPIX / 128), blk, 0, stream>>>(
      H0, 2 * C_PHI, W1T, b1, (void*)H1, HID, NPIX, HID, 2 * C_PHI);
  // layer 2: [M,640] x [640,640], leaky
  gemm_wmma_kernel<1, 0><<<dim3(HID / 128, NPIX / 128), blk, 0, stream>>>(
      H1, HID, W2T, b2, (void*)H2, HID, NPIX, HID, HID);
  // layer 3: [M,640] x [640,384], f32 out
  gemm_wmma_kernel<0, 1><<<dim3((2 * C_LAT) / 128, NPIX / 128), blk, 0, stream>>>(
      H2, HID, W3T, b3, (void*)COND, 2 * C_LAT, NPIX, 2 * C_LAT, HID);

  likelihood_kernel<<<(nx + 255) / 256, 256, 0, stream>>>(COND, out_f32, lik);
}
